// Head_71923522339056
// MI455X (gfx1250) — matrix operand order repaired and checked by reading, not compiled, not run on hardware
//
#include <hip/hip_runtime.h>

#define B_  32
#define T_  2048
#define D_  1024
#define H_  128
#define BT_ (B_ * T_)

typedef __attribute__((ext_vector_type(16))) __bf16 v16bf;
typedef __attribute__((ext_vector_type(8)))  float  v8f;

union Frag { v16bf bf; unsigned int u[8]; };

// Native f32 -> bf16 (RNE) via hardware cvt; avoids 4-op integer emulation.
__device__ __forceinline__ unsigned short f2bf(float f) {
  union { __bf16 h; unsigned short s; } c;
  c.h = (__bf16)f;
  return c.s;
}
__device__ __forceinline__ unsigned int pack_bf16(float lo, float hi) {
  return (unsigned int)f2bf(lo) | ((unsigned int)f2bf(hi) << 16);
}

// Generic pointer -> LDS byte offset (addr[31:0] per CDNA5 aperture mapping).
__device__ __forceinline__ unsigned lds_off32(const void* p) {
  return (unsigned)(unsigned long long)p;
}

// CDNA5 async copy global->LDS, 16B per lane, tracked by ASYNCcnt.
__device__ __forceinline__ void async_ld_b128(unsigned ldsOff, const void* g) {
  asm volatile("global_load_async_to_lds_b128 %0, %1, off"
               :: "v"(ldsOff), "v"(g) : "memory");
}
__device__ __forceinline__ void async_wait0() {
  asm volatile("s_wait_asynccnt 0x0" ::: "memory");
}

// A-matrix 16x32 bf16 fragment (ISA 7.12.2): lane r=lane%16 holds row r.
__device__ __forceinline__ void load_a_frag(Frag& fr, const unsigned short* tile,
                                            int r0, int k0, int lda, int lane) {
  const int r = lane & 15, half = lane >> 4;
  const unsigned int* t32 = (const unsigned int*)tile;
#pragma unroll
  for (int v = 0; v < 8; ++v) {
    const int k = ((v < 4) ? (2 * v) : (2 * v + 8)) + half * 8;
    fr.u[v] = t32[((r0 + r) * lda + k0 + k) >> 1];
  }
}

// B-matrix 32x16 bf16 fragment from TRANSPOSED tile tileT[n][k] (k contiguous).
__device__ __forceinline__ void load_bT_frag(Frag& fr, const unsigned short* tileT,
                                             int n0, int k0, int ldb, int lane) {
  const int n = lane & 15, half = lane >> 4;
  const unsigned int* t32 = (const unsigned int*)tileT;
#pragma unroll
  for (int v = 0; v < 8; ++v) {
    const int k = ((v < 4) ? (2 * v) : (2 * v + 8)) + half * 8;
    fr.u[v] = t32[((n0 + n) * ldb + k0 + k) >> 1];
  }
}

__device__ __forceinline__ v8f wmma_bf16(const Frag& a, const Frag& b, v8f c) {
  return __builtin_amdgcn_wmma_f32_16x16x32_bf16(false, a.bf, false, b.bf,
                                                 (short)0, c, false, false);
}

__device__ __forceinline__ v8f vzero8() {
  v8f z;
#pragma unroll
  for (int e = 0; e < 8; ++e) z[e] = 0.0f;
  return z;
}

// ---------------------------------------------------------------------------
// Kernel 0 (one-time prep): W[D,H] fp32 -> WT[H,D] bf16 (transposed) so the
// GEMM can stage W tiles with straight async global->LDS copies.
// ---------------------------------------------------------------------------
__global__ __launch_bounds__(256) void wprep_kernel(
    const float* __restrict__ W, unsigned short* __restrict__ WT) {
  const int idx = blockIdx.x * 256 + threadIdx.x;  // over D*H
  const int k = idx >> 7;     // H = 128
  const int n = idx & 127;
  WT[(size_t)n * D_ + k] = f2bf(W[idx]);
}

// ---------------------------------------------------------------------------
// Kernel 1 (fused): K/Q/V = x @ {Wk,Wq,Wv} + b, reading x ONCE.
// 256 threads = 8 waves; wave owns a 32x32 patch of each of the 3 outputs.
// W tiles arrive via global_load_async_to_lds_b128 (ASYNCcnt path).
// ---------------------------------------------------------------------------
#define XT_LD 34   // bf16 elements; pairs contiguous, 4B-aligned b32 stores
#define WT_LD 40   // 80B rows: 16B-aligned for B128, dword stride 20 -> no conflicts

__global__ __launch_bounds__(256) void qkv_fused_kernel(
    const float* __restrict__ x, const unsigned short* __restrict__ WT,
    const float* __restrict__ bk, const float* __restrict__ bq,
    const float* __restrict__ bv,
    unsigned short* __restrict__ Kw, unsigned short* __restrict__ Qw,
    unsigned short* __restrict__ Vw) {
  __shared__ alignas(16) unsigned short xt[64 * XT_LD];
  __shared__ alignas(16) unsigned short wt[3][128 * WT_LD];

  const int tid  = threadIdx.x;
  const int lane = tid & 31;
  const int wave = tid >> 5;
  const int wm   = wave & 1;
  const int wn   = wave >> 1;
  const int m0   = blockIdx.x * 64;

  v8f acc[3][2][2];
#pragma unroll
  for (int w = 0; w < 3; ++w)
#pragma unroll
    for (int i = 0; i < 2; ++i)
#pragma unroll
      for (int j = 0; j < 2; ++j) acc[w][i][j] = vzero8();

  for (int k0 = 0; k0 < D_; k0 += 32) {
    __syncthreads();
    // Async-stage the three 128x32 W tiles (8KB each): 512 16B chunks/tile.
#pragma unroll
    for (int w = 0; w < 3; ++w)
#pragma unroll
      for (int j = 0; j < 2; ++j) {
        const int c = tid + j * 256;          // 0..511
        const int n = c >> 2, ch = (c & 3) * 8;
        async_ld_b128(lds_off32(&wt[w][n * WT_LD + ch]),
                      WT + (size_t)w * H_ * D_ + (size_t)n * D_ + k0 + ch);
      }
    // x tile 64x32 fp32 -> bf16: vector loads, packed cvt, b32 LDS stores
    {
      const int r = tid >> 2, cb = (tid & 3) * 8;
      const float4* src4 =
          (const float4*)(x + (size_t)(m0 + r) * D_ + k0 + cb);
      const float4 a = src4[0], b = src4[1];
      unsigned int* dst32 = (unsigned int*)(xt + r * XT_LD + cb);
      dst32[0] = pack_bf16(a.x, a.y);
      dst32[1] = pack_bf16(a.z, a.w);
      dst32[2] = pack_bf16(b.x, b.y);
      dst32[3] = pack_bf16(b.z, b.w);
    }
    async_wait0();
    __syncthreads();

    Frag a0, a1;
    load_a_frag(a0, xt, wm * 32,      0, XT_LD, lane);
    load_a_frag(a1, xt, wm * 32 + 16, 0, XT_LD, lane);
#pragma unroll
    for (int w = 0; w < 3; ++w)
#pragma unroll
      for (int nf = 0; nf < 2; ++nf) {
        Frag bfr;
        load_bT_frag(bfr, wt[w], wn * 32 + nf * 16, 0, WT_LD, lane);
        acc[w][0][nf] = wmma_bf16(a0, bfr, acc[w][0][nf]);
        acc[w][1][nf] = wmma_bf16(a1, bfr, acc[w][1][nf]);
      }
  }

  const int n = lane & 15, half = lane >> 4;
#pragma unroll
  for (int w = 0; w < 3; ++w) {
    const float* bias = (w == 0) ? bk : (w == 1) ? bq : bv;
    unsigned short* out = (w == 0) ? Kw : (w == 1) ? Qw : Vw;
#pragma unroll
    for (int nf = 0; nf < 2; ++nf) {
      const int col = wn * 32 + nf * 16 + n;
      const float bb = bias[col];
#pragma unroll
      for (int mf = 0; mf < 2; ++mf)
#pragma unroll
        for (int v = 0; v < 8; ++v) {
          const int row = m0 + wm * 32 + mf * 16 + v + half * 8;
          out[(size_t)row * H_ + col] = f2bf(acc[w][mf][nf][v] + bb);
        }
    }
  }
}

// ---------------------------------------------------------------------------
// Kernel 2: flash causal attention, wei = K @ Q^T (reference quirk).
// Q s-tile staged via async global->LDS; V transposed manually in LDS.
// ---------------------------------------------------------------------------
#define QT_LD 136  // 272B rows: 16B aligned, dword stride 68 (=4 mod 64) -> no conflicts
#define VT_LD 34
#define PT_LD 34

__global__ __launch_bounds__(128) void attn_kernel(
    const unsigned short* __restrict__ Kg, const unsigned short* __restrict__ Qg,
    const unsigned short* __restrict__ Vg, float* __restrict__ out) {
  __shared__ alignas(16) unsigned short qt[32 * QT_LD];
  __shared__ alignas(16) unsigned short vt[128 * VT_LD];
  __shared__ alignas(16) unsigned short pt[4][16 * PT_LD];

  const int tid  = threadIdx.x;
  const int lane = tid & 31;
  const int wave = tid >> 5;            // 4 waves
  const int t0   = blockIdx.x * 64;
  const int tw   = t0 + wave * 16;
  const size_t base = (size_t)blockIdx.y * T_ * H_;

  // This wave's 16 K rows (16x128) as 4 A-fragments, gathered from global
  Frag afr[4];
  {
    const int r = lane & 15, half = lane >> 4;
    const unsigned int* Kp = (const unsigned int*)(Kg + base + (size_t)tw * H_);
#pragma unroll
    for (int f = 0; f < 4; ++f)
#pragma unroll
      for (int v = 0; v < 8; ++v) {
        const int k = f * 32 + ((v < 4) ? (2 * v) : (2 * v + 8)) + half * 8;
        afr[f].u[v] = Kp[(r * H_ + k) >> 1];
      }
  }

  v8f o[8];
  float m[8], l[8];
#pragma unroll
  for (int i = 0; i < 8; ++i) { o[i] = vzero8(); m[i] = -1e30f; l[i] = 0.0f; }
  const float scale = 0.08838834764831845f;  // 1/sqrt(H)

  const int sEnd = t0 + 64;
  for (int s0 = 0; s0 < sEnd; s0 += 32) {
    __syncthreads();
    // Q tile (32x128 bf16 = 8KB): async copy, 512 16B chunks / 128 threads
#pragma unroll
    for (int j = 0; j < 4; ++j) {
      const int c = tid + j * 128;            // 0..511
      const int s = c >> 4, ch = (c & 15) * 8;
      async_ld_b128(lds_off32(&qt[s * QT_LD + ch]),
                    Qg + base + (size_t)(s0 + s) * H_ + ch);
    }
    // V tile transposed (pairs must run along s for the B fragment)
    {
      const int r = tid >> 2, cb = (tid & 3) * 32;
      const unsigned short* vs = Vg + base + (size_t)(s0 + r) * H_ + cb;
#pragma unroll
      for (int i = 0; i < 32; ++i) vt[(cb + i) * VT_LD + r] = vs[i];
    }
    async_wait0();
    __syncthreads();

    if (s0 <= tw + 15) {  // wave-uniform skip of fully-masked s-blocks
      v8f sc[2];
      sc[0] = vzero8(); sc[1] = vzero8();
#pragma unroll
      for (int nf = 0; nf < 2; ++nf)
#pragma unroll
        for (int kf = 0; kf < 4; ++kf) {
          Frag bq;
          load_bT_frag(bq, qt, nf * 16, kf * 32, QT_LD, lane);
          sc[nf] = wmma_bf16(afr[kf], bq, sc[nf]);
        }

      const int n = lane & 15, half = lane >> 4;
      unsigned short* myP = pt[wave];
#pragma unroll
      for (int v = 0; v < 8; ++v) {
        const int row = tw + v + half * 8;
        float x0 = (s0 + n      <= row) ? sc[0][v] * scale : -1e30f;
        float x1 = (s0 + 16 + n <= row) ? sc[1][v] * scale : -1e30f;
        float rmax = fmaxf(x0, x1);
#pragma unroll
        for (int off = 8; off >= 1; off >>= 1)
          rmax = fmaxf(rmax, __shfl_xor(rmax, off, 32));
        const float mnew = fmaxf(m[v], rmax);
        const float p0 = __expf(x0 - mnew);
        const float p1 = __expf(x1 - mnew);
        float rsum = p0 + p1;
#pragma unroll
        for (int off = 8; off >= 1; off >>= 1)
          rsum += __shfl_xor(rsum, off, 32);
        const float alpha = __expf(m[v] - mnew);
        l[v] = l[v] * alpha + rsum;
        m[v] = mnew;
#pragma unroll
        for (int nf = 0; nf < 8; ++nf) o[nf][v] *= alpha;
        const int prow = v + half * 8;
        myP[prow * PT_LD + n]      = f2bf(p0);
        myP[prow * PT_LD + 16 + n] = f2bf(p1);
      }
      asm volatile("s_wait_dscnt 0x0" ::: "memory");  // P is wave-private LDS

      Frag ap;
      load_a_frag(ap, myP, 0, 0, PT_LD, lane);
#pragma unroll
      for (int nf = 0; nf < 8; ++nf) {
        Frag bv;
        load_bT_frag(bv, vt, nf * 16, 0, VT_LD, lane);
        o[nf] = wmma_bf16(ap, bv, o[nf]);
      }
    }
  }

  const int n = lane & 15, half = lane >> 4;
#pragma unroll
  for (int v = 0; v < 8; ++v) {
    const float inv = 1.0f / l[v];
    const int row = tw + v + half * 8;
    float* dst = out + base + (size_t)row * H_;
#pragma unroll
    for (int nf = 0; nf < 8; ++nf) dst[nf * 16 + n] = o[nf][v] * inv;
  }
}

extern "C" void kernel_launch(void* const* d_in, const int* in_sizes, int n_in,
                              void* d_out, int out_size, void* d_ws, size_t ws_size,
                              hipStream_t stream) {
  (void)in_sizes; (void)n_in; (void)out_size; (void)ws_size;
  const float* x  = (const float*)d_in[0];
  const float* Wk = (const float*)d_in[1];
  const float* bk = (const float*)d_in[2];
  const float* Wq = (const float*)d_in[3];
  const float* bq = (const float*)d_in[4];
  const float* Wv = (const float*)d_in[5];
  const float* bv = (const float*)d_in[6];

  unsigned short* Kw = (unsigned short*)d_ws;            // bf16 K [BT,H]
  unsigned short* Qw = Kw + (size_t)BT_ * H_;            // bf16 Q
  unsigned short* Vw = Qw + (size_t)BT_ * H_;            // bf16 V
  unsigned short* WT = Vw + (size_t)BT_ * H_;            // bf16 [3][H][D]

  dim3 gprep(D_ * H_ / 256);
  wprep_kernel<<<gprep, 256, 0, stream>>>(Wk, WT);
  wprep_kernel<<<gprep, 256, 0, stream>>>(Wq, WT + (size_t)H_ * D_);
  wprep_kernel<<<gprep, 256, 0, stream>>>(Wv, WT + 2 * (size_t)H_ * D_);

  dim3 gproj(BT_ / 64);
  qkv_fused_kernel<<<gproj, 256, 0, stream>>>(x, WT, bk, bq, bv, Kw, Qw, Vw);

  dim3 gattn(T_ / 64, B_);
  attn_kernel<<<gattn, 128, 0, stream>>>(Kw, Qw, Vw, (float*)d_out);
}
